// MFModel_12781822673306
// MI455X (gfx1250) — compile-verified, matrix operand order checked by reading, not run on hardware
//
#include <hip/hip_runtime.h>
#include <hip/hip_bf16.h>

// ---------------------------------------------------------------------------
// out[u,i] = dot(user_hiddens[user_ids[u]], item_hiddens[item_ids[i]])
// 4096x4096 output, K=64, fp32 in/out.
// Strategy: bf16 split-precision (hi+lo) WMMA GEMM, fp32 accumulate.
// ---------------------------------------------------------------------------

typedef __attribute__((ext_vector_type(16))) __bf16 v16bf;
typedef __attribute__((ext_vector_type(8)))  float  v8f;

#define DDIM        64     // hidden dim (K)
#define BU          128    // users per block
#define BI          64     // items per block
#define LDS_STRIDE  72     // elements per LDS row: 144B = 9*16B (aligned, conflict-free)
#define BLOCK_THREADS 256

__device__ __forceinline__ unsigned short f32_to_bf16_rne(float f) {
    unsigned int u = __float_as_uint(f);
    u += 0x7fffu + ((u >> 16) & 1u);   // round-to-nearest-even (data is finite, [0,1))
    return (unsigned short)(u >> 16);
}

union FragU {
    uint4 q[2];
    v16bf v;
};

// Load one 16x32 bf16 fragment (this lane's slice) from an LDS row.
// CDNA5 16-bit A layout: lane group g holds K = {8g..8g+7} in VGPR0-3 and
// K = {16+8g..16+8g+7} in VGPR4-7 (row = lane%16). B mirrors with lanes = N.
__device__ __forceinline__ v16bf load_frag(const unsigned short* __restrict__ rowbase,
                                           int g, int kchunk) {
    FragU f;
    f.q[0] = *(const uint4*)(rowbase + kchunk +      8 * g);
    f.q[1] = *(const uint4*)(rowbase + kchunk + 16 + 8 * g);
    return f.v;
}

__device__ __forceinline__ v8f wmma_bf16(v16bf a, v16bf b, v8f c) {
    return __builtin_amdgcn_wmma_f32_16x16x32_bf16(
        /*neg_a=*/false, a, /*neg_b=*/false, b,
        /*c_mod=*/(short)0, c, /*reuse_a=*/false, /*reuse_b=*/false);
}

__global__ __launch_bounds__(BLOCK_THREADS)
void mf_dot_wmma_kernel(const float* __restrict__ user_h,
                        const float* __restrict__ item_h,
                        const int*   __restrict__ user_ids,
                        const int*   __restrict__ item_ids,
                        float*       __restrict__ out,
                        int out_w) {
    __shared__ unsigned short sUhi[BU * LDS_STRIDE];
    __shared__ unsigned short sUlo[BU * LDS_STRIDE];
    __shared__ unsigned short sIhi[BI * LDS_STRIDE];
    __shared__ unsigned short sIlo[BI * LDS_STRIDE];

    const int tid = threadIdx.x;
    const int ub  = blockIdx.y * BU;   // user-row base of this block's macro tile
    const int ib  = blockIdx.x * BI;   // item-col base

    // ---- Stage gathered user rows -> LDS as bf16 hi/lo (float4 granularity) ----
    for (int i = tid; i < BU * (DDIM / 4); i += BLOCK_THREADS) {
        const int r  = i >> 4;            // 16 float4 per row
        const int c4 = (i & 15) << 2;     // element column
        const long long row = (long long)user_ids[ub + r];
        const float4 x = *(const float4*)(user_h + row * DDIM + c4);
        const float xs[4] = {x.x, x.y, x.z, x.w};
        unsigned short h[4], l[4];
#pragma unroll
        for (int j = 0; j < 4; ++j) {
            h[j] = f32_to_bf16_rne(xs[j]);
            const float hf = __uint_as_float((unsigned int)h[j] << 16);
            l[j] = f32_to_bf16_rne(xs[j] - hf);
        }
        const uint2 hp = make_uint2((unsigned)h[0] | ((unsigned)h[1] << 16),
                                    (unsigned)h[2] | ((unsigned)h[3] << 16));
        const uint2 lp = make_uint2((unsigned)l[0] | ((unsigned)l[1] << 16),
                                    (unsigned)l[2] | ((unsigned)l[3] << 16));
        *(uint2*)&sUhi[r * LDS_STRIDE + c4] = hp;
        *(uint2*)&sUlo[r * LDS_STRIDE + c4] = lp;
    }

    // ---- Stage gathered item rows -> LDS as bf16 hi/lo ----
    for (int i = tid; i < BI * (DDIM / 4); i += BLOCK_THREADS) {
        const int r  = i >> 4;
        const int c4 = (i & 15) << 2;
        const long long row = (long long)item_ids[ib + r];
        const float4 x = *(const float4*)(item_h + row * DDIM + c4);
        const float xs[4] = {x.x, x.y, x.z, x.w};
        unsigned short h[4], l[4];
#pragma unroll
        for (int j = 0; j < 4; ++j) {
            h[j] = f32_to_bf16_rne(xs[j]);
            const float hf = __uint_as_float((unsigned int)h[j] << 16);
            l[j] = f32_to_bf16_rne(xs[j] - hf);
        }
        const uint2 hp = make_uint2((unsigned)h[0] | ((unsigned)h[1] << 16),
                                    (unsigned)h[2] | ((unsigned)h[3] << 16));
        const uint2 lp = make_uint2((unsigned)l[0] | ((unsigned)l[1] << 16),
                                    (unsigned)l[2] | ((unsigned)l[3] << 16));
        *(uint2*)&sIhi[r * LDS_STRIDE + c4] = hp;
        *(uint2*)&sIlo[r * LDS_STRIDE + c4] = lp;
    }

    __syncthreads();

    // ---- WMMA compute: each wave owns one 16-user strip, 4 item tiles ----
    const int wave = tid >> 5;
    const int lane = tid & 31;
    const int g    = lane >> 4;   // lane group (K-subset select)
    const int m    = lane & 15;   // row (A) / column (B) index

    const int arow = wave * 16 + m;
    const unsigned short* uHi = sUhi + arow * LDS_STRIDE;
    const unsigned short* uLo = sUlo + arow * LDS_STRIDE;

    const v16bf aH0 = load_frag(uHi, g, 0);
    const v16bf aH1 = load_frag(uHi, g, 32);
    const v16bf aL0 = load_frag(uLo, g, 0);
    const v16bf aL1 = load_frag(uLo, g, 32);

#pragma unroll
    for (int tn = 0; tn < BI / 16; ++tn) {
        const int brow = tn * 16 + m;
        const unsigned short* iHi = sIhi + brow * LDS_STRIDE;
        const unsigned short* iLo = sIlo + brow * LDS_STRIDE;

        const v16bf bH0 = load_frag(iHi, g, 0);
        const v16bf bH1 = load_frag(iHi, g, 32);
        const v16bf bL0 = load_frag(iLo, g, 0);
        const v16bf bL1 = load_frag(iLo, g, 32);

        v8f c = {};
        // hi*hi (K=0..31, K=32..63)
        c = wmma_bf16(aH0, bH0, c);
        c = wmma_bf16(aH1, bH1, c);
        // lo*hi
        c = wmma_bf16(aL0, bH0, c);
        c = wmma_bf16(aL1, bH1, c);
        // hi*lo
        c = wmma_bf16(aH0, bL0, c);
        c = wmma_bf16(aH1, bL1, c);

        // C/D layout: VGPR r -> M = r + 8*g (lanes 0-15 / 16-31), N = lane%16
        const int ocol = ib + tn * 16 + m;
#pragma unroll
        for (int r = 0; r < 8; ++r) {
            const int orow = ub + wave * 16 + r + 8 * g;
            __builtin_nontemporal_store(c[r], out + (size_t)orow * (size_t)out_w + ocol);
        }
    }
}

extern "C" void kernel_launch(void* const* d_in, const int* in_sizes, int n_in,
                              void* d_out, int out_size, void* d_ws, size_t ws_size,
                              hipStream_t stream) {
    const float* user_h   = (const float*)d_in[0];
    const float* item_h   = (const float*)d_in[1];
    const int*   user_ids = (const int*)d_in[2];
    const int*   item_ids = (const int*)d_in[3];
    float*       out      = (float*)d_out;

    const int batch_users = in_sizes[2];   // 4096
    const int batch_items = in_sizes[3];   // 4096

    dim3 grid(batch_items / BI, batch_users / BU);
    mf_dot_wmma_kernel<<<grid, BLOCK_THREADS, 0, stream>>>(
        user_h, item_h, user_ids, item_ids, out, batch_items);
}
